// LSTMCellHidden_11819749999298
// MI455X (gfx1250) — matrix operand order repaired and checked by reading, not compiled, not run on hardware
//
#include <hip/hip_runtime.h>
#include <hip/hip_bf16.h>
#include <stdint.h>

// ---------------------------------------------------------------------------
// Fused LSTM-cell (hidden-driven gates) for MI455X / gfx1250.
//   i = sigmoid(h @ Wi + bi); f = sigmoid(h @ Wf + bf)
//   g = tanh   (h @ Wg + bg); o = sigmoid(h @ Wo + bo)
//   c' = f*c + i*g ; h' = o*tanh(c')
//
// Pipeline:
//   pass 1: W[k][n] fp32 -> WT[g][n][k] bf16 (d_ws, 8 MB)      [one-time]
//   pass 2: hprev fp32 -> bf16 (d_ws+8MB, 16 MB)               [one-time]
//   pass 3: fused 4-gate bf16 WMMA GEMM + LSTM epilogue.
//           Double-buffered LDS staged with global_load_async_to_lds_b128
//           (ASYNCcnt), one barrier per K-step, 16 WMMAs per K-step.
// ---------------------------------------------------------------------------

typedef __attribute__((ext_vector_type(16))) __bf16 bf16x16;
typedef __attribute__((ext_vector_type(8)))  __bf16 bf16x8;
typedef __attribute__((ext_vector_type(8)))  float  f32x8;
typedef __attribute__((ext_vector_type(4)))  float  f32x4;

#define B_DIM 8192
#define HD    1024
#define CD    1024
#define BM    128
#define BN    64
#define BK    32
#define KT    (HD / BK)
#define NGATE 4

__device__ __forceinline__ __bf16 f2bf(float f) {
  uint32_t u = __builtin_bit_cast(uint32_t, f);
  u += 0x7FFFu + ((u >> 16) & 1u);       // round-to-nearest-even
  uint16_t h = (uint16_t)(u >> 16);
  return __builtin_bit_cast(__bf16, h);
}

__device__ __forceinline__ float sigmoidf_(float x) {
  return 1.0f / (1.0f + __expf(-x));
}
__device__ __forceinline__ float tanhf_(float x) {
  float e = __expf(-2.0f * x);
  return (1.0f - e) / (1.0f + e);
}

// CDNA5 async global->LDS copy, 16 bytes per lane, tracked by ASYNCcnt.
__device__ __forceinline__ void async_b128(uint32_t lds_addr, const void* gaddr) {
  asm volatile("global_load_async_to_lds_b128 %0, %1, off"
               :: "v"(lds_addr), "v"(gaddr)
               : "memory");
}
__device__ __forceinline__ void wait_async0() {
  asm volatile("s_wait_asynccnt 0x0" ::: "memory");
}

// ---- one-time: W[k][n] fp32 -> WT[g][n][k] bf16 (tiled transpose) ----
__global__ void convert_wt_kernel(const float* __restrict__ w0,
                                  const float* __restrict__ w1,
                                  const float* __restrict__ w2,
                                  const float* __restrict__ w3,
                                  __bf16* __restrict__ wt) {
  __shared__ __bf16 tile[32][33];
  const float* w = (blockIdx.z == 0) ? w0 :
                   (blockIdx.z == 1) ? w1 :
                   (blockIdx.z == 2) ? w2 : w3;
  __bf16* out = wt + (size_t)blockIdx.z * HD * CD;
  const int tx = threadIdx.x, ty = threadIdx.y;
  const int n = blockIdx.x * 32 + tx;
#pragma unroll
  for (int j = 0; j < 32; j += 8) {
    int k = blockIdx.y * 32 + ty + j;
    tile[ty + j][tx] = f2bf(w[(size_t)k * CD + n]);   // coalesced fp32 read
  }
  __syncthreads();
#pragma unroll
  for (int j = 0; j < 32; j += 8) {
    int n2 = blockIdx.x * 32 + ty + j;
    int k2 = blockIdx.y * 32 + tx;
    out[(size_t)n2 * HD + k2] = tile[tx][ty + j];     // coalesced bf16 write
  }
}

// ---- one-time: hprev fp32 -> bf16 (elementwise, bandwidth bound) ----
__global__ void convert_h_kernel(const float* __restrict__ h,
                                 __bf16* __restrict__ hb) {
  size_t i = ((size_t)blockIdx.x * blockDim.x + threadIdx.x) * 8;
  f32x4 a = *(const f32x4*)(h + i);
  f32x4 b = *(const f32x4*)(h + i + 4);
  bf16x8 o;
#pragma unroll
  for (int j = 0; j < 4; ++j) {
    o[j]     = f2bf(a[j]);
    o[4 + j] = f2bf(b[j]);
  }
  *(bf16x8*)(hb + i) = o;
}

// ---- fused 4-gate WMMA GEMM + LSTM epilogue ----
__global__ void __launch_bounds__(256)
lstm_fused_kernel(const __bf16* __restrict__ hb,     // [B][HD] bf16
                  const float* __restrict__ cprev,
                  const __bf16* __restrict__ wt,     // [4][CD][HD] bf16
                  const float* __restrict__ b_hi,
                  const float* __restrict__ b_hf,
                  const float* __restrict__ b_hg,
                  const float* __restrict__ b_ho,
                  float* __restrict__ out) {
  // Double-buffered LDS: A row-major [BM][BK]; B N-major [gate][BN][BK]
  __shared__ __align__(16) __bf16 sA[2][BM][BK];          // 16 KB
  __shared__ __align__(16) __bf16 sB[2][NGATE][BN][BK];   // 32 KB

  const int tid   = threadIdx.x;
  const int lane  = tid & 31;
  const int wid   = tid >> 5;        // 8 waves
  const int waveM = wid >> 1;        // 0..3 -> 32-row strip
  const int waveN = wid & 1;         // 0..1 -> 32-col strip
  const int rowBlk = blockIdx.y * BM;
  const int colBlk = blockIdx.x * BN;

  const int halfSel = lane >> 4;     // lane 0..15 vs 16..31
  const int llow    = lane & 15;

  f32x8 acc[NGATE][2][2] = {};       // 4 gates x (2x2) 16x16 f32 tiles

  // async-stage one K tile (A: 2 chunks/thread, B: 1 chunk/thread/gate)
  auto stage = [&](int buf, int k0) {
#pragma unroll
    for (int j = 0; j < 2; ++j) {
      int c   = tid + j * 256;       // 512 x 16B chunks
      int row = c >> 2;
      int kc  = (c & 3) * 8;
      const __bf16* g = hb + (size_t)(rowBlk + row) * HD + (k0 + kc);
      async_b128((uint32_t)(uintptr_t)&sA[buf][row][kc], g);
    }
#pragma unroll
    for (int gt = 0; gt < NGATE; ++gt) {
      int n  = tid >> 2;
      int kc = (tid & 3) * 8;
      const __bf16* g = wt + ((size_t)gt * CD + (colBlk + n)) * HD + (k0 + kc);
      async_b128((uint32_t)(uintptr_t)&sB[buf][gt][n][kc], g);
    }
  };

  stage(0, 0);
  wait_async0();
  __syncthreads();

  for (int kt = 0; kt < KT; ++kt) {
    const int cur = kt & 1;
    if (kt + 1 < KT) stage(cur ^ 1, (kt + 1) * BK);   // overlap with compute

    // ---- A fragments (ISA 16-bit 16x32 layout) ----
    // lanes 0-15: M=lane, K 0..7 & 16..23 ; lanes 16-31: K 8..15 & 24..31
    bf16x16 afrag[2];
#pragma unroll
    for (int mi = 0; mi < 2; ++mi) {
      const __bf16* base = &sA[cur][waveM * 32 + mi * 16 + llow][halfSel * 8];
      bf16x8 lo = *(const bf16x8*)(base);
      bf16x8 hi = *(const bf16x8*)(base + 16);
      afrag[mi] = __builtin_shufflevector(lo, hi, 0, 1, 2, 3, 4, 5, 6, 7,
                                          8, 9, 10, 11, 12, 13, 14, 15);
    }
    // ---- B fragments, all gates up-front so dscnt waits batch ----
    // lanes 0-15: N=lane, K 0..7 & 16..23 ; lanes 16-31: N=lane-16, K 8..15 & 24..31
    bf16x16 bfrag[NGATE][2];
#pragma unroll
    for (int g = 0; g < NGATE; ++g) {
#pragma unroll
      for (int ni = 0; ni < 2; ++ni) {
        const __bf16* base =
            &sB[cur][g][waveN * 32 + ni * 16 + llow][halfSel * 8];
        bf16x8 lo = *(const bf16x8*)(base);
        bf16x8 hi = *(const bf16x8*)(base + 16);
        bfrag[g][ni] = __builtin_shufflevector(lo, hi, 0, 1, 2, 3, 4, 5, 6, 7,
                                               8, 9, 10, 11, 12, 13, 14, 15);
      }
    }
    // ---- 16 WMMAs per K-step ----
#pragma unroll
    for (int g = 0; g < NGATE; ++g)
#pragma unroll
      for (int ni = 0; ni < 2; ++ni)
#pragma unroll
        for (int mi = 0; mi < 2; ++mi)
          acc[g][mi][ni] = __builtin_amdgcn_wmma_f32_16x16x32_bf16(
              false, afrag[mi], false, bfrag[g][ni], (short)0,
              acc[g][mi][ni], false, false);

    wait_async0();       // next-tile async copies landed
    __syncthreads();     // all waves done reading cur / writing nxt
  }

  // ---- fused LSTM epilogue (C/D layout: VGPR v -> M = halfSel*8+v, N = llow)
  float bI[2], bF[2], bG[2], bO[2];
#pragma unroll
  for (int ni = 0; ni < 2; ++ni) {
    int col = colBlk + waveN * 32 + ni * 16 + llow;
    bI[ni] = b_hi[col];
    bF[ni] = b_hf[col];
    bG[ni] = b_hg[col];
    bO[ni] = b_ho[col];
  }
  float* outH = out;
  float* outC = out + (size_t)B_DIM * CD;

#pragma unroll
  for (int mi = 0; mi < 2; ++mi) {
#pragma unroll
    for (int ni = 0; ni < 2; ++ni) {
      const int col = colBlk + waveN * 32 + ni * 16 + llow;
#pragma unroll
      for (int v = 0; v < 8; ++v) {
        const int row = rowBlk + waveM * 32 + mi * 16 + halfSel * 8 + v;
        const size_t idx = (size_t)row * CD + col;
        float iv = sigmoidf_(acc[0][mi][ni][v] + bI[ni]);
        float fv = sigmoidf_(acc[1][mi][ni][v] + bF[ni]);
        float gv = tanhf_  (acc[2][mi][ni][v] + bG[ni]);
        float ov = sigmoidf_(acc[3][mi][ni][v] + bO[ni]);
        float cn = fv * cprev[idx] + iv * gv;
        outH[idx] = ov * tanhf_(cn);
        outC[idx] = cn;
      }
    }
  }
}

extern "C" void kernel_launch(void* const* d_in, const int* in_sizes, int n_in,
                              void* d_out, int out_size, void* d_ws, size_t ws_size,
                              hipStream_t stream) {
  (void)in_sizes; (void)n_in; (void)out_size; (void)ws_size;
  const float* hprev = (const float*)d_in[0];
  const float* cprev = (const float*)d_in[1];
  const float* w_hi  = (const float*)d_in[2];
  const float* b_hi  = (const float*)d_in[3];
  const float* w_hf  = (const float*)d_in[4];
  const float* b_hf  = (const float*)d_in[5];
  const float* w_hg  = (const float*)d_in[6];
  const float* b_hg  = (const float*)d_in[7];
  const float* w_ho  = (const float*)d_in[8];
  const float* b_ho  = (const float*)d_in[9];

  // d_ws layout: [0, 8MB)  bf16 transposed weights  [4][CD][HD]
  //              [8MB, 24MB) bf16 hprev             [B][HD]
  __bf16* wt = (__bf16*)d_ws;
  __bf16* hb = (__bf16*)((char*)d_ws + (size_t)NGATE * HD * CD * sizeof(__bf16));

  dim3 cb(32, 8);
  dim3 cg(CD / 32, HD / 32, 4);
  convert_wt_kernel<<<cg, cb, 0, stream>>>(w_hi, w_hf, w_hg, w_ho, wt);

  convert_h_kernel<<<(B_DIM * HD) / (256 * 8), 256, 0, stream>>>(hprev, hb);

  dim3 mb(256);
  dim3 mg(CD / BN, B_DIM / BM);   // 16 x 64 workgroups
  lstm_fused_kernel<<<mg, mb, 0, stream>>>(hb, cprev, wt,
                                           b_hi, b_hf, b_hg, b_ho,
                                           (float*)d_out);
}
